// SQAE_1752346656836
// MI455X (gfx1250) — compile-verified, matrix operand order checked by reading, not run on hardware
//
// SQAE (VQ-VAE-ish) forward for gfx1250 (MI455X).
// ~635 GFLOP of GEMM vs ~15us of HBM traffic at 23.3 TB/s -> compute-bound:
// bf16 WMMA (v_wmma_f32_16x16x32_bf16) with f32 accumulation everywhere.
// GEMMs templated on (N,K) -> immediate-offset stores/loads; LDS double
// buffered, one barrier per k-step; staging uses GLOBAL_LOAD_ASYNC_TO_LDS
// (ASYNCcnt) when the toolchain exposes it, else global->reg->LDS fallback.
// Quantizer fuses GEMM + argmax (avoids a 1 GB score matrix); codebook
// norms are stored as reciprocals so the score is a multiply, not a divide.
#include <hip/hip_runtime.h>
#include <hip/hip_bf16.h>
#include <stdint.h>

typedef __attribute__((ext_vector_type(16))) __bf16 v16bf;
typedef __attribute__((ext_vector_type(8)))  float  v8f;

#if __has_builtin(__builtin_amdgcn_global_load_async_to_lds_b128) && \
    __has_builtin(__builtin_amdgcn_s_wait_asynccnt)
#define HAVE_ASYNC_LDS 1
#else
#define HAVE_ASYNC_LDS 0
#endif

#if HAVE_ASYNC_LDS
typedef int v4i __attribute__((vector_size(16)));
typedef __attribute__((address_space(1))) v4i g_v4i;   // global (AS1)
typedef __attribute__((address_space(3))) v4i l_v4i;   // LDS (AS3)
__device__ __forceinline__ void async_ld16(const unsigned short* g, unsigned short* l) {
  __builtin_amdgcn_global_load_async_to_lds_b128(
      (g_v4i*)(unsigned short*)g, (l_v4i*)l, 0, 0);
}
#endif

struct Frag32B { uint4 lo, hi; };

__device__ __forceinline__ unsigned short f32_to_bf16_bits(float f) {
  unsigned int u = __float_as_uint(f);
  u += 0x7FFFu + ((u >> 16) & 1u);          // round-to-nearest-even
  return (unsigned short)(u >> 16);
}

// Build a v16bf WMMA fragment from two 16B LDS reads.
// Lane L: row/col = L&15, khalf = L>>4. Elements 0..7 <- K = 8*khalf..+7,
// elements 8..15 <- K = 16+8*khalf..+7 (per CDNA5 16-bit A/B layout).
__device__ __forceinline__ v16bf load_frag(const unsigned short* p) {
  Frag32B f;
  f.lo = *(const uint4*)p;
  f.hi = *(const uint4*)(p + 16);
  return __builtin_bit_cast(v16bf, f);
}

// ---------------------------------------------------------------------------
// GEMM: C[M,N] = A[M,K](bf16) * W[N,K]^T(bf16) + bias, C f32.
// Block 128x128, BK=32, 8 waves (4 in M x 2 in N), wave tile 32x64
// (2x4 WMMA tiles = 8 wmma / k-step). Double-buffered LDS, 1 barrier/step.
// ---------------------------------------------------------------------------
#define GBM 128
#define GBN 128
#define GBK 32
#define GLD 40   // padded LDS row stride (ushorts); 80B rows stay 16B aligned

template <int N, int K>
__global__ __launch_bounds__(256) void gemm_bias(
    const unsigned short* __restrict__ A,
    const unsigned short* __restrict__ W,
    const float* __restrict__ bias,
    float* __restrict__ C)
{
  __shared__ __align__(16) unsigned short sA[2][GBM * GLD];
  __shared__ __align__(16) unsigned short sB[2][GBN * GLD];

  const int tid  = threadIdx.x;
  const int lane = tid & 31;
  const int wave = tid >> 5;
  const int bm = blockIdx.y * GBM;
  const int bn = blockIdx.x * GBN;
  const int wm = (wave >> 1) * 32;   // 0,32,64,96
  const int wn = (wave & 1) * 64;    // 0,64
  const int lm = lane & 15;
  const int kh = lane >> 4;

  // Staging chunk coordinates (2 x 16B chunks per thread for each of A,B).
  const int r0 = tid >> 2,          c0 = (tid & 3) * 8;
  const int r1 = (tid + 256) >> 2,  c1 = ((tid + 256) & 3) * 8;

  v8f acc[2][4];
#pragma unroll
  for (int a = 0; a < 2; ++a)
#pragma unroll
    for (int b = 0; b < 4; ++b)
#pragma unroll
      for (int e = 0; e < 8; ++e) acc[a][b][e] = 0.0f;

#if HAVE_ASYNC_LDS
  auto stage = [&](int k0, int buf) {
    async_ld16(A + (size_t)(bm + r0) * K + k0 + c0, &sA[buf][r0 * GLD + c0]);
    async_ld16(A + (size_t)(bm + r1) * K + k0 + c1, &sA[buf][r1 * GLD + c1]);
    async_ld16(W + (size_t)(bn + r0) * K + k0 + c0, &sB[buf][r0 * GLD + c0]);
    async_ld16(W + (size_t)(bn + r1) * K + k0 + c1, &sB[buf][r1 * GLD + c1]);
  };
  stage(0, 0);
#else
  uint4 ra0, ra1, rb0, rb1;
  auto gload = [&](int k0) {
    ra0 = *(const uint4*)(A + (size_t)(bm + r0) * K + k0 + c0);
    ra1 = *(const uint4*)(A + (size_t)(bm + r1) * K + k0 + c1);
    rb0 = *(const uint4*)(W + (size_t)(bn + r0) * K + k0 + c0);
    rb1 = *(const uint4*)(W + (size_t)(bn + r1) * K + k0 + c1);
  };
  gload(0);
#endif

  constexpr int NK = K / GBK;
  for (int kk = 0; kk < NK; ++kk) {
    const int buf = kk & 1;
#if HAVE_ASYNC_LDS
    __builtin_amdgcn_s_wait_asynccnt(0);   // step-kk DMA done (this wave)
    __syncthreads();                       // ... and visible to all waves
    if (kk + 1 < NK) stage((kk + 1) * GBK, buf ^ 1);  // overlaps WMMA below
#else
    *(uint4*)&sA[buf][r0 * GLD + c0] = ra0;
    *(uint4*)&sA[buf][r1 * GLD + c1] = ra1;
    *(uint4*)&sB[buf][r0 * GLD + c0] = rb0;
    *(uint4*)&sB[buf][r1 * GLD + c1] = rb1;
    __syncthreads();
    if (kk + 1 < NK) gload((kk + 1) * GBK);
#endif

    v16bf af[2], bfb[4];
#pragma unroll
    for (int t = 0; t < 2; ++t)
      af[t] = load_frag(&sA[buf][(wm + t * 16 + lm) * GLD + 8 * kh]);
#pragma unroll
    for (int t = 0; t < 4; ++t)
      bfb[t] = load_frag(&sB[buf][(wn + t * 16 + lm) * GLD + 8 * kh]);
#pragma unroll
    for (int tm = 0; tm < 2; ++tm)
#pragma unroll
      for (int tn = 0; tn < 4; ++tn)
        acc[tm][tn] = __builtin_amdgcn_wmma_f32_16x16x32_bf16(
            false, af[tm], false, bfb[tn], (short)0, acc[tm][tn], false, false);
    // No trailing barrier: next step writes the other buffer; reads of this
    // buffer retire (dscnt waits precede the WMMAs) before the next signal.
  }

  // Epilogue: constexpr N -> base pointer per tm + immediate store offsets.
#pragma unroll
  for (int tm = 0; tm < 2; ++tm) {
    const int row0 = bm + wm + tm * 16 + 8 * kh;
    float* cp = C + (size_t)row0 * N + bn + wn + lm;
#pragma unroll
    for (int tn = 0; tn < 4; ++tn) {
      float bv = bias[bn + wn + tn * 16 + lm];
#pragma unroll
      for (int i = 0; i < 8; ++i)
        cp[(size_t)i * N + tn * 16] = acc[tm][tn][i] + bv;
    }
  }
}

// ---------------------------------------------------------------------------
// LayerNorm over D=512 rows; one wave per row (16 f32/lane).
// MODE 0: y = LN(x)               -> outF + outB
// MODE 1: y = tanh(LN(x))         -> outB
// MODE 2: y = tanh(LN(x) + hprev) -> outF + outB
// ---------------------------------------------------------------------------
template <int MODE>
__global__ __launch_bounds__(256) void ln_kernel(
    const float* __restrict__ X,
    const float* __restrict__ g, const float* __restrict__ be,
    const float* __restrict__ hprev,
    float* __restrict__ outF, unsigned short* __restrict__ outB,
    int D)
{
  int row  = blockIdx.x * 8 + (threadIdx.x >> 5);
  int lane = threadIdx.x & 31;
  const int PER = D >> 5;       // 16 for D=512
  float v[16];
  const float* xr = X + (size_t)row * D;

  float s = 0.0f;
#pragma unroll
  for (int i = 0; i < 16; ++i)
    if (i < PER) { v[i] = xr[lane + 32 * i]; s += v[i]; }
#pragma unroll
  for (int off = 16; off >= 1; off >>= 1) s += __shfl_xor(s, off, 32);
  float mean = s / (float)D;

  float q = 0.0f;
#pragma unroll
  for (int i = 0; i < 16; ++i)
    if (i < PER) { float d = v[i] - mean; q += d * d; }
#pragma unroll
  for (int off = 16; off >= 1; off >>= 1) q += __shfl_xor(q, off, 32);
  float inv = rsqrtf(q / (float)D + 1e-5f);

#pragma unroll
  for (int i = 0; i < 16; ++i) {
    if (i < PER) {
      int c = lane + 32 * i;
      float y = (v[i] - mean) * inv * g[c] + be[c];
      if (MODE == 1) y = tanhf(y);
      if (MODE == 2) y = tanhf(y + hprev[(size_t)row * D + c]);
      if (MODE != 1) outF[(size_t)row * D + c] = y;
      outB[(size_t)row * D + c] = f32_to_bf16_bits(y);
    }
  }
}

// ---------------------------------------------------------------------------
// Codebook prep: rn[k] = 1/||emb[k]|| (reciprocal!), emb -> bf16.
// One wave per row (D=256).
// ---------------------------------------------------------------------------
__global__ __launch_bounds__(256) void emb_prep(
    const float* __restrict__ emb, unsigned short* __restrict__ eb,
    float* __restrict__ rn, int De)
{
  int row  = blockIdx.x * 8 + (threadIdx.x >> 5);
  int lane = threadIdx.x & 31;
  float s = 0.0f;
  for (int i = 0; i < De / 32; ++i) {
    float f = emb[(size_t)row * De + lane + 32 * i];
    eb[(size_t)row * De + lane + 32 * i] = f32_to_bf16_bits(f);
    s += f * f;
  }
#pragma unroll
  for (int off = 16; off >= 1; off >>= 1) s += __shfl_xor(s, off, 32);
  if (lane == 0) rn[row] = rsqrtf(s);   // argmax is scale-invariant per row
}

__global__ void cvt_bf16(const float* __restrict__ in,
                         unsigned short* __restrict__ out, size_t n) {
  size_t i = blockIdx.x * (size_t)blockDim.x + threadIdx.x;
  size_t st = (size_t)gridDim.x * blockDim.x;
  for (; i < n; i += st) out[i] = f32_to_bf16_bits(in[i]);
}

// ---------------------------------------------------------------------------
// Fused quantizer: argmax_k (z.emb_k * rn_k) via WMMA, then gather.
// Block: 64 z-rows staged once in LDS (read-only thereafter -> barrier-free);
// loop all Kc codes in 128-wide tiles (8 waves = 2M x 4N), double-buffered sB
// (one barrier per k-step, async staging incl. cross-tile prefetch). Packed
// (sortable-float, ~idx) u64 atomicMax gives argmin-lowest-index ties.
// ---------------------------------------------------------------------------
#define QBM 64
#define QBN 128
#define QSA 264   // 256 + 8 pad
#define QLDB 40

__device__ __forceinline__ unsigned long long packScore(float v, unsigned int idx) {
  unsigned int b = __float_as_uint(v);
  b = (b & 0x80000000u) ? ~b : (b | 0x80000000u);
  return ((unsigned long long)b << 32) | (unsigned long long)(0xFFFFFFFFu - idx);
}

__global__ __launch_bounds__(256) void quant_argmax(
    const unsigned short* __restrict__ Zb,   // [B,256] bf16
    const unsigned short* __restrict__ Eb,   // [Kc,256] bf16
    const float* __restrict__ rn,            // [Kc] reciprocal norms
    const float* __restrict__ embF,          // [Kc,256] f32
    float* __restrict__ zq,                  // [B,256]
    int* __restrict__ idxOut,                // [B]
    unsigned short* __restrict__ zqb,        // [B,256] bf16
    int Kc)
{
  __shared__ __align__(16) unsigned short sA[QBM * QSA];
  __shared__ __align__(16) unsigned short sB[2][QBN * QLDB];
  __shared__ unsigned long long best[QBM];

  const int tid  = threadIdx.x;
  const int lane = tid & 31;
  const int wave = tid >> 5;
  const int bm = blockIdx.x * QBM;
  const int wm = (wave >> 2) * 32;   // 0,32
  const int wn = (wave & 3) * 32;    // 0..96
  const int lm = lane & 15;
  const int kh = lane >> 4;

  const int r0 = tid >> 2,          c0 = (tid & 3) * 8;
  const int r1 = (tid + 256) >> 2,  c1 = ((tid + 256) & 3) * 8;

#if HAVE_ASYNC_LDS
  auto stageB = [&](int bnn, int k0, int buf) {
    async_ld16(Eb + (size_t)(bnn + r0) * 256 + k0 + c0, &sB[buf][r0 * QLDB + c0]);
    async_ld16(Eb + (size_t)(bnn + r1) * 256 + k0 + c1, &sB[buf][r1 * QLDB + c1]);
  };
  // Stage the whole 64x256 z tile (2048 16B chunks, 8/thread) via DMA too.
#pragma unroll
  for (int i = 0; i < 8; ++i) {
    int c  = tid + i * 256;
    int r  = c >> 5;
    int cc = c & 31;
    async_ld16(Zb + (size_t)(bm + r) * 256 + cc * 8, &sA[r * QSA + cc * 8]);
  }
  stageB(0, 0, 0);
#else
#pragma unroll
  for (int i = 0; i < 8; ++i) {
    int c  = tid + i * 256;
    int r  = c >> 5;
    int cc = c & 31;
    *(uint4*)&sA[r * QSA + cc * 8] =
        *(const uint4*)(Zb + (size_t)(bm + r) * 256 + cc * 8);
  }
#endif
  if (tid < QBM) best[tid] = 0ull;

  float bestV[2][8];
  unsigned int bestI[2][8];
#pragma unroll
  for (int tm = 0; tm < 2; ++tm)
#pragma unroll
    for (int i = 0; i < 8; ++i) { bestV[tm][i] = -3.402823466e38f; bestI[tm][i] = 0u; }

  for (int bn = 0; bn < Kc; bn += QBN) {
    v8f acc[2][2];
#pragma unroll
    for (int a = 0; a < 2; ++a)
#pragma unroll
      for (int b = 0; b < 2; ++b)
#pragma unroll
        for (int e = 0; e < 8; ++e) acc[a][b][e] = 0.0f;

#if !HAVE_ASYNC_LDS
    uint4 rb0 = *(const uint4*)(Eb + (size_t)(bn + r0) * 256 + c0);
    uint4 rb1 = *(const uint4*)(Eb + (size_t)(bn + r1) * 256 + c1);
#endif

#pragma unroll
    for (int kk = 0; kk < 8; ++kk) {
      const int buf = kk & 1;
#if HAVE_ASYNC_LDS
      __builtin_amdgcn_s_wait_asynccnt(0);
      __syncthreads();
      if (kk < 7)               stageB(bn, (kk + 1) * 32, buf ^ 1);
      else if (bn + QBN < Kc)   stageB(bn + QBN, 0, buf ^ 1);
#else
      *(uint4*)&sB[buf][r0 * QLDB + c0] = rb0;
      *(uint4*)&sB[buf][r1 * QLDB + c1] = rb1;
      __syncthreads();
      if (kk < 7) {
        rb0 = *(const uint4*)(Eb + (size_t)(bn + r0) * 256 + (kk + 1) * 32 + c0);
        rb1 = *(const uint4*)(Eb + (size_t)(bn + r1) * 256 + (kk + 1) * 32 + c1);
      }
#endif

      v16bf af[2], bfb[2];
#pragma unroll
      for (int t = 0; t < 2; ++t) {
        af[t]  = load_frag(&sA[(wm + t * 16 + lm) * QSA + kk * 32 + 8 * kh]);
        bfb[t] = load_frag(&sB[buf][(wn + t * 16 + lm) * QLDB + 8 * kh]);
      }
#pragma unroll
      for (int tm = 0; tm < 2; ++tm)
#pragma unroll
        for (int tn = 0; tn < 2; ++tn)
          acc[tm][tn] = __builtin_amdgcn_wmma_f32_16x16x32_bf16(
              false, af[tm], false, bfb[tn], (short)0, acc[tm][tn], false, false);
    }

#pragma unroll
    for (int tn = 0; tn < 2; ++tn) {
      int col = bn + wn + tn * 16 + lm;
      float rcp = rn[col];                 // reciprocal norm: multiply only
#pragma unroll
      for (int tm = 0; tm < 2; ++tm)
#pragma unroll
        for (int i = 0; i < 8; ++i) {
          float sc = acc[tm][tn][i] * rcp;
          if (sc > bestV[tm][i]) { bestV[tm][i] = sc; bestI[tm][i] = (unsigned)col; }
        }
    }
  }

#pragma unroll
  for (int tm = 0; tm < 2; ++tm)
#pragma unroll
    for (int i = 0; i < 8; ++i) {
      int r = wm + tm * 16 + i + 8 * kh;
      atomicMax(&best[r], packScore(bestV[tm][i], bestI[tm][i]));
    }
  __syncthreads();

  // Gather: 4 threads per row copy emb[idx] (f32 + bf16) and write idx.
  int r = tid >> 2, q = tid & 3;
  unsigned long long pk = best[r];
  unsigned int idx = 0xFFFFFFFFu - (unsigned int)(pk & 0xFFFFFFFFull);
  int grow = bm + r;
  if (q == 0) idxOut[grow] = (int)idx;
  for (int j = 0; j < 64; ++j) {
    int c = q * 64 + j;
    float v = embF[(size_t)idx * 256 + c];
    zq[(size_t)grow * 256 + c]  = v;
    zqb[(size_t)grow * 256 + c] = f32_to_bf16_bits(v);
  }
}

// ---------------------------------------------------------------------------
extern "C" void kernel_launch(void* const* d_in, const int* in_sizes, int n_in,
                              void* d_out, int out_size, void* d_ws, size_t ws_size,
                              hipStream_t stream) {
  (void)n_in; (void)out_size; (void)ws_size;
  const float* x       = (const float*)d_in[0];
  const float* enc_w0  = (const float*)d_in[1];
  const float* enc_b0  = (const float*)d_in[2];
  const float* enc_g0  = (const float*)d_in[3];
  const float* enc_be0 = (const float*)d_in[4];
  const float* enc_rw  = (const float*)d_in[5];
  const float* enc_rb  = (const float*)d_in[6];
  const float* enc_rg  = (const float*)d_in[7];
  const float* enc_rbe = (const float*)d_in[8];
  const float* enc_wf  = (const float*)d_in[9];
  const float* enc_bf  = (const float*)d_in[10];
  const float* emb     = (const float*)d_in[11];
  const float* dec_w0  = (const float*)d_in[12];
  const float* dec_b0  = (const float*)d_in[13];
  const float* dec_g0  = (const float*)d_in[14];
  const float* dec_be0 = (const float*)d_in[15];
  const float* dec_rw  = (const float*)d_in[16];
  const float* dec_rb  = (const float*)d_in[17];
  const float* dec_rg  = (const float*)d_in[18];
  const float* dec_rbe = (const float*)d_in[19];
  const float* dec_wf  = (const float*)d_in[20];
  const float* dec_bf  = (const float*)d_in[21];

  const int Din = in_sizes[21];              // 1024
  const int Dh  = in_sizes[2];               // 512
  const int De  = in_sizes[10];              // 256
  const int B   = in_sizes[0] / Din;         // 32768
  const int Kc  = in_sizes[11] / De;         // 8192
  const int NR  = in_sizes[5] / (Dh * Dh);   // 6

  // Workspace carve-up (256B aligned).
  char* ws = (char*)d_ws;
  size_t off = 0;
  auto alloc = [&](size_t bytes) -> char* {
    char* p = ws + off;
    off = (off + bytes + 255) & ~(size_t)255;
    return p;
  };
  unsigned short* wb_ew0 = (unsigned short*)alloc((size_t)Dh * Din * 2);
  unsigned short* wb_erw = (unsigned short*)alloc((size_t)NR * Dh * Dh * 2);
  unsigned short* wb_ewf = (unsigned short*)alloc((size_t)De * Dh * 2);
  unsigned short* wb_dw0 = (unsigned short*)alloc((size_t)Dh * De * 2);
  unsigned short* wb_drw = (unsigned short*)alloc((size_t)NR * Dh * Dh * 2);
  unsigned short* wb_dwf = (unsigned short*)alloc((size_t)Din * Dh * 2);
  unsigned short* embB   = (unsigned short*)alloc((size_t)Kc * De * 2);
  float*          rn     = (float*)alloc((size_t)Kc * 4);
  unsigned short* xb     = (unsigned short*)alloc((size_t)B * Din * 2);
  unsigned short* hb     = (unsigned short*)alloc((size_t)B * Dh * 2);
  unsigned short* zb     = (unsigned short*)alloc((size_t)B * De * 2);
  unsigned short* zqb    = (unsigned short*)alloc((size_t)B * De * 2);
  float*          h      = (float*)alloc((size_t)B * Dh * 4);
  float*          t      = (float*)alloc((size_t)B * Dh * 4);
  // xb is dead after the first encoder GEMM; reuse it for ob (bf16 tanh(LN)).
  unsigned short* ob = xb;

  // Output layout: x_recon[B,Din] | z[B,De] | z_q[B,De] | idx[B] (int bits).
  float* xr   = (float*)d_out;
  float* z    = xr + (size_t)B * Din;
  float* zq   = z + (size_t)B * De;
  int*   idxO = (int*)(zq + (size_t)B * De);

  auto gemm = [&](const unsigned short* A, const unsigned short* W,
                  const float* bias, float* C, int M, int n, int k) {
    dim3 g(n / GBN, M / GBM);
    if      (n ==  512 && k == 1024) gemm_bias< 512,1024><<<g,256,0,stream>>>(A,W,bias,C);
    else if (n ==  512 && k ==  512) gemm_bias< 512, 512><<<g,256,0,stream>>>(A,W,bias,C);
    else if (n ==  256 && k ==  512) gemm_bias< 256, 512><<<g,256,0,stream>>>(A,W,bias,C);
    else if (n ==  512 && k ==  256) gemm_bias< 512, 256><<<g,256,0,stream>>>(A,W,bias,C);
    else if (n == 1024 && k ==  512) gemm_bias<1024, 512><<<g,256,0,stream>>>(A,W,bias,C);
  };
  auto ln = [&](const float* X, const float* gg, const float* bb,
                const float* hp, float* oF, unsigned short* oB, int mode) {
    if (mode == 0)      ln_kernel<0><<<B/8,256,0,stream>>>(X,gg,bb,hp,oF,oB,Dh);
    else if (mode == 1) ln_kernel<1><<<B/8,256,0,stream>>>(X,gg,bb,hp,oF,oB,Dh);
    else                ln_kernel<2><<<B/8,256,0,stream>>>(X,gg,bb,hp,oF,oB,Dh);
  };

  // --- precision conversion (weights tiny; fully L2-resident) ---
  cvt_bf16<<<2048, 256, 0, stream>>>(x, xb, (size_t)B * Din);
  cvt_bf16<<<512, 256, 0, stream>>>(enc_w0, wb_ew0, (size_t)Dh * Din);
  cvt_bf16<<<512, 256, 0, stream>>>(enc_rw, wb_erw, (size_t)NR * Dh * Dh);
  cvt_bf16<<<256, 256, 0, stream>>>(enc_wf, wb_ewf, (size_t)De * Dh);
  cvt_bf16<<<256, 256, 0, stream>>>(dec_w0, wb_dw0, (size_t)Dh * De);
  cvt_bf16<<<512, 256, 0, stream>>>(dec_rw, wb_drw, (size_t)NR * Dh * Dh);
  cvt_bf16<<<512, 256, 0, stream>>>(dec_wf, wb_dwf, (size_t)Din * Dh);
  emb_prep<<<Kc / 8, 256, 0, stream>>>(emb, embB, rn, De);

  // --- encoder ---
  gemm(xb, wb_ew0, enc_b0, t, B, Dh, Din);
  ln(t, enc_g0, enc_be0, nullptr, h, hb, 0);
  for (int i = 0; i < NR; ++i) {
    const unsigned short* W  = wb_erw + (size_t)i * Dh * Dh;
    const float* bb = enc_rb + (size_t)i * Dh;
    const float* gg = enc_rg + (size_t)i * Dh;
    const float* be = enc_rbe + (size_t)i * Dh;
    gemm(hb, W, bb, t, B, Dh, Dh);
    ln(t, gg, be, nullptr, nullptr, ob, 1);
    gemm(ob, W, bb, t, B, Dh, Dh);
    ln(t, gg, be, h, h, hb, 2);
  }
  gemm(hb, wb_ewf, enc_bf, z, B, De, Dh);

  // --- quantizer (straight-through: decoder input == z_q) ---
  cvt_bf16<<<2048, 256, 0, stream>>>(z, zb, (size_t)B * De);
  quant_argmax<<<B / QBM, 256, 0, stream>>>(zb, embB, rn, emb, zq, idxO, zqb, Kc);

  // --- decoder ---
  gemm(zqb, wb_dw0, dec_b0, t, B, Dh, De);
  ln(t, dec_g0, dec_be0, nullptr, h, hb, 0);
  for (int i = 0; i < NR; ++i) {
    const unsigned short* W  = wb_drw + (size_t)i * Dh * Dh;
    const float* bb = dec_rb + (size_t)i * Dh;
    const float* gg = dec_rg + (size_t)i * Dh;
    const float* be = dec_rbe + (size_t)i * Dh;
    gemm(hb, W, bb, t, B, Dh, Dh);
    ln(t, gg, be, nullptr, nullptr, ob, 1);
    gemm(ob, W, bb, t, B, Dh, Dh);
    ln(t, gg, be, h, h, hb, 2);
  }
  gemm(hb, wb_dwf, dec_bf, xr, B, Din, Dh);
}
